// ASymplecticR4_NN_84009560310416
// MI455X (gfx1250) — matrix operand order repaired and checked by reading, not compiled, not run on hardware
//
#include <hip/hip_runtime.h>

// ---------------------------------------------------------------------------
// TinyMLP (2->32->32->1, sigmoid) analytic input-gradient, B = 4,194,304.
//
// Roofline: 128 MiB of stream traffic (~5.5us @ 23.3 TB/s) vs ~17 GFLOP of
// 32x32 batched GEMM work. Both GEMMs (H1@W2 forward, G2@W2^T backward) run
// on the matrix pipe with v_wmma_f32_16x16x32_bf16 (bf16 in, fp32 accum).
// Sigmoids use v_tanh_f32 (1 TRANS op each). The 0.5 pre-scale that tanh
// needs is folded into the layer-1 weights and the forward B-tile (0.5*W2,
// exact in bf16), so WMMA emits z/2 directly; the resulting 2x on dY is
// folded into the final output FMA. sigma'(z1) is computed in stage A from
// the live tanh register (2 VALU) instead of re-derived from bf16 h1.
// All staging LDS is wave-private -> wave-local s_wait_dscnt fences only.
// ---------------------------------------------------------------------------

typedef __attribute__((ext_vector_type(16))) __bf16 v16bf;
typedef __attribute__((ext_vector_type(8)))  __bf16 v8bf;
typedef __attribute__((ext_vector_type(8)))  float  v8f;
typedef __attribute__((ext_vector_type(4)))  float  v4f;

#define NW 8    // waves per block
#define HP 40   // padded LDS row for bf16 tiles (32 + 8 pad halves = 80B)
#define FP 36   // padded LDS row for f32 tiles  (32 + 4 pad floats = 144B)

// Wave-local LDS fence: staging tiles are wave-private, so we only need the
// wave's own DS ops complete (DS pipe is in-order per wave) plus a compiler
// reordering fence. No cross-wave barrier.
__device__ __forceinline__ void wave_lds_fence() {
  asm volatile("s_wait_dscnt 0x0" ::: "memory");
  __builtin_amdgcn_wave_barrier();
}

__global__ __launch_bounds__(256)
void tinymlp_inputgrad_kernel(const float* __restrict__ x1, const float* __restrict__ x2,
                              const float* __restrict__ y1, const float* __restrict__ y2,
                              const float* __restrict__ W1, const float* __restrict__ b1,
                              const float* __restrict__ W2, const float* __restrict__ b2,
                              const float* __restrict__ W3, float* __restrict__ out,
                              int ntiles) {
  // Per-wave private staging tiles (32 samples x 32 hidden), padded rows.
  __shared__ alignas(16) __bf16 h1s[NW][32][HP];   // h1 = sigmoid(Y W1 + b1), bf16
  __shared__ alignas(16) float  d1s[NW][32][FP];   // sigma'(z1), f32
  __shared__ alignas(16) __bf16 g2s[NW][32][HP];   // W3 .* sigma'(z2), bf16
  __shared__ alignas(16) float  g1s[NW][32][FP];   // dH1 .* sigma'(z1), f32
  // Block-shared weights (read-only after the prologue barrier).
  __shared__ alignas(16) __bf16 w2s [32][HP];      // W2 row-major (backward B, unscaled)
  __shared__ alignas(16) __bf16 w2ts[32][HP];      // 0.5*W2 col-major (forward B, pre-scaled)
  __shared__ alignas(16) float4 w1s[32];           // 0.5*(W1[0][k], W1[1][k], b1[k])
  __shared__ float b2s_[32];                       // 0.5*b2
  __shared__ float w3s_[32];

  const int tid  = threadIdx.x;
  const int wid  = tid >> 5;    // wave id in block
  const int lane = tid & 31;
  const int lh   = lane >> 4;   // lane half (0/1)
  const int ln   = lane & 15;

  // ---- one-time weight preload ----
  for (int i = tid; i < 1024; i += 256) {
    int r = i >> 5, c = i & 31;
    float v = W2[i];
    w2s[r][c]  = (__bf16)v;            // backward: W2^T as B, unscaled
    w2ts[c][r] = (__bf16)(0.5f * v);   // forward: 0.5*W2 as B (exact bf16 scale)
  }
  if (tid < 32) {
    w1s[tid]  = make_float4(0.5f * W1[tid], 0.5f * W1[32 + tid], 0.5f * b1[tid], 0.0f);
    b2s_[tid] = 0.5f * b2[tid];
    w3s_[tid] = W3[tid];
  }
  __syncthreads();   // only cross-wave sync in the kernel

  // ---- loop-invariant per-lane state ----
  float cinit[2], ql[2];
  cinit[0] = b2s_[ln];          cinit[1] = b2s_[ln + 16];    // 0.5*b2[n]
  ql[0] = 0.25f * w3s_[ln];     ql[1] = 0.25f * w3s_[ln + 16]; // q = 0.25*W3[n]

  const int K0a = lh * 8;    // A-matrix per-lane K chunks: [K0a,K0a+8) U [K0a+16,K0a+24)
  const int K0b = lh * 16;   // unused: B tiles load the same K0a chunks as A

  // Hoist both B-matrix tiles (0.5*W2 forward, W2^T backward) into VGPRs.
  v16bf Bf[2], Bb[2];
#pragma unroll
  for (int nt = 0; nt < 2; ++nt) {
    int n = ln + 16 * nt;
    v8bf lo = *(const v8bf*)&w2ts[n][K0a];       // Bf(k,n) = 0.5*W2[k][n]
    v8bf hi = *(const v8bf*)&w2ts[n][K0a + 16];
    Bf[nt] = __builtin_shufflevector(lo, hi, 0,1,2,3,4,5,6,7,8,9,10,11,12,13,14,15);
    lo = *(const v8bf*)&w2s[n][K0a];             // Bb(k,n) = W2[n][k]  (= W2^T)
    hi = *(const v8bf*)&w2s[n][K0a + 16];
    Bb[nt] = __builtin_shufflevector(lo, hi, 0,1,2,3,4,5,6,7,8,9,10,11,12,13,14,15);
  }

  // ---- grid-stride over 256-sample tiles (uniform trip count per block) ----
  for (int t = blockIdx.x; t < ntiles; t += gridDim.x) {
    const int m = t * 256 + wid * 32 + lane;     // this lane's sample
    float x1v = x1[m], x2v = x2[m], y1v = y1[m], y2v = y2[m];

    int tnext = t + gridDim.x;                   // uniform guard -> EXEC stays full
    if (tnext < ntiles) {
      int pm = tnext * 256 + wid * 32 + lane;
      __builtin_prefetch(&x1[pm], 0, 1);
      __builtin_prefetch(&x2[pm], 0, 1);
      __builtin_prefetch(&y1[pm], 0, 1);
      __builtin_prefetch(&y2[pm], 0, 1);
    }

    // ---- layer 1 (K=2, VALU): zh = z/2 via pre-scaled weights ----
    v8bf hv[4];
    v4f  spv[8];
#pragma unroll
    for (int k = 0; k < 32; ++k) {
      float4 wk = w1s[k];                                   // 0.5*(W1a, W1b, b1)
      float zh = fmaf(y1v, wk.x, fmaf(y2v, wk.y, wk.z));    // z1/2
#if __has_builtin(__builtin_amdgcn_tanhf)
      float tt = __builtin_amdgcn_tanhf(zh);                // 1 TRANS
      float h  = fmaf(0.5f, tt, 0.5f);                      // sigmoid(z1)
      float sp = fmaf(-0.25f * tt, tt, 0.25f);              // sigma'(z1)
#else
      float e  = __builtin_amdgcn_exp2f(-zh * 2.8853900817779268f); // 2*log2(e)
      float h  = __builtin_amdgcn_rcpf(1.0f + e);
      float sp = h * (1.0f - h);
#endif
      hv[k >> 3][k & 7]  = (__bf16)h;
      spv[k >> 2][k & 3] = sp;
    }
#pragma unroll
    for (int qd = 0; qd < 4; ++qd)
      *(v8bf*)&h1s[wid][lane][qd * 8] = hv[qd];
#pragma unroll
    for (int qd = 0; qd < 8; ++qd)
      *(v4f*)&d1s[wid][lane][qd * 4] = spv[qd];
    wave_lds_fence();

    // ---- GEMM1: c = (H1 @ W2 + b2)/2  (4x v_wmma_f32_16x16x32_bf16) ----
    v16bf a[2];
#pragma unroll
    for (int mt = 0; mt < 2; ++mt) {
      int row = 16 * mt + ln;                    // A: M = ln
      v8bf lo = *(const v8bf*)&h1s[wid][row][K0a];
      v8bf hi = *(const v8bf*)&h1s[wid][row][K0a + 16];
      a[mt] = __builtin_shufflevector(lo, hi, 0,1,2,3,4,5,6,7,8,9,10,11,12,13,14,15);
    }
    v8f c[2][2];
#pragma unroll
    for (int mt = 0; mt < 2; ++mt)
#pragma unroll
      for (int nt = 0; nt < 2; ++nt) {
        v8f ci;
#pragma unroll
        for (int v = 0; v < 8; ++v) ci[v] = cinit[nt];   // 0.5*b2[n] splat
        c[mt][nt] = __builtin_amdgcn_wmma_f32_16x16x32_bf16(
            false, a[mt], false, Bf[nt], (short)0, ci, false, false);
      }

    // ---- backward seed: g2 = W3[n]*sigma'(z2) = q - q*t^2, t = tanh(z2/2) ----
    // C layout: VGPR v holds (row = 16*mt + v + 8*lh, col = ln + 16*nt)
#pragma unroll
    for (int mt = 0; mt < 2; ++mt)
#pragma unroll
      for (int nt = 0; nt < 2; ++nt)
#pragma unroll
        for (int v = 0; v < 8; ++v) {
          float cc = c[mt][nt][v];                         // z2/2 already
#if __has_builtin(__builtin_amdgcn_tanhf)
          float tt = __builtin_amdgcn_tanhf(cc);           // 1 TRANS
          float g  = fmaf(-ql[nt] * tt, tt, ql[nt]);
#else
          float e  = __builtin_amdgcn_exp2f(-cc * 2.8853900817779268f);
          float s  = __builtin_amdgcn_rcpf(1.0f + e);
          float g  = 4.0f * ql[nt] * s * (1.0f - s);
#endif
          g2s[wid][16 * mt + v + 8 * lh][ln + 16 * nt] = (__bf16)g;
        }
    wave_lds_fence();

    // ---- GEMM2: dH1 = G2 @ W2^T  (4x v_wmma_f32_16x16x32_bf16, C = 0) ----
    v16bf a2[2];
#pragma unroll
    for (int mt = 0; mt < 2; ++mt) {
      int row = 16 * mt + ln;
      v8bf lo = *(const v8bf*)&g2s[wid][row][K0a];
      v8bf hi = *(const v8bf*)&g2s[wid][row][K0a + 16];
      a2[mt] = __builtin_shufflevector(lo, hi, 0,1,2,3,4,5,6,7,8,9,10,11,12,13,14,15);
    }
    v8f d[2][2];
#pragma unroll
    for (int mt = 0; mt < 2; ++mt)
#pragma unroll
      for (int nt = 0; nt < 2; ++nt) {
        v8f z0 = {0.f, 0.f, 0.f, 0.f, 0.f, 0.f, 0.f, 0.f};
        d[mt][nt] = __builtin_amdgcn_wmma_f32_16x16x32_bf16(
            false, a2[mt], false, Bb[nt], (short)0, z0, false, false);
      }

    // ---- g1 = dH1 .* sigma'(z1), transposed back to sample-major ----
#pragma unroll
    for (int mt = 0; mt < 2; ++mt)
#pragma unroll
      for (int nt = 0; nt < 2; ++nt)
#pragma unroll
        for (int v = 0; v < 8; ++v) {
          int r = 16 * mt + v + 8 * lh;
          int k = ln + 16 * nt;
          g1s[wid][r][k] = d[mt][nt][v] * d1s[wid][r][k];  // ld + mul + st
        }
    wave_lds_fence();

    // ---- dY = 2 * (0.5*W1) . g1  (length-32 dot per sample) ----
    float dY0 = 0.0f, dY1 = 0.0f;
#pragma unroll
    for (int qd = 0; qd < 8; ++qd) {
      float4 gq  = *(const float4*)&g1s[wid][lane][qd * 4];
      float4 w0  = w1s[qd * 4 + 0];
      float4 w1v = w1s[qd * 4 + 1];
      float4 w2v = w1s[qd * 4 + 2];
      float4 w3v = w1s[qd * 4 + 3];
      dY0 = fmaf(gq.x, w0.x,  dY0); dY1 = fmaf(gq.x, w0.y,  dY1);
      dY0 = fmaf(gq.y, w1v.x, dY0); dY1 = fmaf(gq.y, w1v.y, dY1);
      dY0 = fmaf(gq.z, w2v.x, dY0); dY1 = fmaf(gq.z, w2v.y, dY1);
      dY0 = fmaf(gq.w, w3v.x, dY0); dY1 = fmaf(gq.w, w3v.y, dY1);
    }

    // undo the 0.5 pre-scale of W1 with a free FMA
    float4 o = make_float4(fmaf(2.0f, dY0, x1v), fmaf(2.0f, dY1, x2v), y1v, y2v);
    ((float4*)out)[m] = o;   // coalesced global_store_b128
  }
}

extern "C" void kernel_launch(void* const* d_in, const int* in_sizes, int n_in,
                              void* d_out, int out_size, void* d_ws, size_t ws_size,
                              hipStream_t stream) {
  const float* x1 = (const float*)d_in[0];
  const float* x2 = (const float*)d_in[1];
  const float* y1 = (const float*)d_in[2];
  const float* y2 = (const float*)d_in[3];
  const float* W1 = (const float*)d_in[4];  // [2,32]
  const float* b1 = (const float*)d_in[5];  // [32]
  const float* W2 = (const float*)d_in[6];  // [32,32]
  const float* b2 = (const float*)d_in[7];  // [32]
  const float* W3 = (const float*)d_in[8];  // [32,1]
  // d_in[9] = b3: constant offset, drops out of the input gradient.

  const int B = in_sizes[0];                // 4,194,304
  const int ntiles = B / 256;               // 16384 tiles of 256 samples
  int grid = ntiles < 2048 ? (ntiles > 0 ? ntiles : 1) : 2048;

  tinymlp_inputgrad_kernel<<<grid, 256, 0, stream>>>(
      x1, x2, y1, y2, W1, b1, W2, b2, W3, (float*)d_out, ntiles);
}
